// MultiHeadSelfAttention_9002251453083
// MI455X (gfx1250) — matrix operand order repaired and checked by reading, not compiled, not run on hardware
//
#include <hip/hip_runtime.h>

typedef __attribute__((ext_vector_type(16))) _Float16 v16h;
typedef __attribute__((ext_vector_type(8)))  _Float16 v8h;
typedef __attribute__((ext_vector_type(8)))  float    v8f;
typedef __attribute__((ext_vector_type(4)))  unsigned int u32x4;
typedef __attribute__((ext_vector_type(8)))  int i32x8;
typedef __attribute__((ext_vector_type(4)))  int i32x4;

#define BATCH 4
#define SEQ   2048
#define DM    1024
#define NH    16
#define HD    64
#define N3    3072
#define NEG_INF (-3.0e38f)

// ---------------------------------------------------------------- converts
__global__ void f32_to_f16_kernel(const float* __restrict__ in,
                                  _Float16* __restrict__ out, int n) {
  int i = blockIdx.x * blockDim.x + threadIdx.x;
  int stride = gridDim.x * blockDim.x;
  for (; i < n; i += stride) out[i] = (_Float16)in[i];
}

// ---------------------------------------------------------------- RoPE (in-place on f16 qkv, q & k sections)
__global__ void rope_kernel(_Float16* __restrict__ qkv,
                            const int* __restrict__ pos) {
  int idx = blockIdx.x * blockDim.x + threadIdx.x;
  int p  = idx & 31;  int t = idx >> 5;   // pair index 0..31
  int h  = t & 15;    t >>= 4;            // head
  int qk = t & 1;     t >>= 1;            // 0=q, 1=k
  int s  = t & 2047;  t >>= 11;           // seq
  int b  = t;                             // batch
  size_t base = ((size_t)(b * SEQ + s)) * N3 + qk * DM + h * HD + 2 * p;
  float fp  = (float)pos[s];
  float inv = powf(10000.0f, -(2.0f * (float)p) / 64.0f);
  float ang = fp * inv;
  float c = cosf(ang), sn = sinf(ang);
  float x1 = (float)qkv[base], x2 = (float)qkv[base + 1];
  qkv[base]     = (_Float16)(x1 * c - x2 * sn);
  qkv[base + 1] = (_Float16)(x2 * c + x1 * sn);
}

// ---------------------------------------------------------------- TDM 2-D tile load: global -> LDS
// Descriptor layout per CDNA5 ISA (groups 0/1; groups 2/3 zero for 2-D).
__device__ __forceinline__ void tdm_load_2d(unsigned lds_off, const void* gaddr,
                                            unsigned tile_d0, unsigned tile_d1,
                                            unsigned long long stride0_elems) {
  unsigned long long ga = (unsigned long long)(uintptr_t)gaddr;
  u32x4 g0;
  g0[0] = 1u;                                                // count=1 (valid), user mode
  g0[1] = lds_off;                                           // lds_addr (bytes)
  g0[2] = (unsigned)(ga & 0xFFFFFFFFu);                      // global_addr[31:0]
  g0[3] = (unsigned)((ga >> 32) & 0x1FFFFFFu) | (2u << 30);  // global_addr[56:32] | type=2
  i32x8 g1;
  g1[0] = 0x00010000;                                        // data_size=1 (2B), wg_mask=0
  g1[1] = (int)((tile_d0 & 0xFFFFu) << 16);                  // tensor_dim0 = tile_d0
  g1[2] = (int)((tile_d0 >> 16) | ((tile_d1 & 0xFFFFu) << 16)); // tensor_dim1 = tile_d1
  g1[3] = (int)((tile_d1 >> 16) | (tile_d0 << 16));          // tile_dim0
  g1[4] = (int)(tile_d1 & 0xFFFFu);                          // tile_dim1 (tile_dim2=0)
  g1[5] = (int)(stride0_elems & 0xFFFFFFFFull);              // tensor_dim0_stride lo
  g1[6] = (int)((stride0_elems >> 32) & 0xFFFFull);          // stride hi (dim1_stride=0)
  g1[7] = 0;
  i32x4 z4 = {};
#if __clang_major__ >= 23
  i32x8 z8 = {};
  __builtin_amdgcn_tensor_load_to_lds(g0, g1, z4, z4, z8, 0);
#else
  __builtin_amdgcn_tensor_load_to_lds(g0, g1, z4, z4, 0);
#endif
}

// ---------------------------------------------------------------- WMMA GEMM with TDM-fed LDS double buffering
// C[M][N] = A[M][K] * W[N][K]^T.  Block = 128(M) x 64(N); 8 waves each own 16(M) x 64(N).
__global__ __launch_bounds__(256) void gemm_wmma_tdm(
    const _Float16* __restrict__ A, const _Float16* __restrict__ W,
    _Float16* __restrict__ Ch, float* __restrict__ Cf,
    int M, int N, int K) {
  __shared__ __align__(64) _Float16 Abuf[2][128 * 32];
  __shared__ __align__(64) _Float16 Wbuf[2][64 * 32];

  const int lane = threadIdx.x & 31;
  const int lh   = lane & 15;
  const int hi   = lane >> 4;
  const int w    = threadIdx.x >> 5;
  const int tiles_n = N >> 6;
  const int m0 = (blockIdx.x / tiles_n) * 128;
  const int n0 = (blockIdx.x % tiles_n) * 64;
  const int nk = K >> 5;
  const bool issuer = (threadIdx.x < 32);  // wave 0 drives the TDM

  auto issue = [&](int kt) {
    const int k0 = kt << 5;
    tdm_load_2d((unsigned)(uintptr_t)&Abuf[kt & 1][0],
                A + (size_t)m0 * K + k0, 32u, 128u, (unsigned long long)K);
    tdm_load_2d((unsigned)(uintptr_t)&Wbuf[kt & 1][0],
                W + (size_t)n0 * K + k0, 32u, 64u, (unsigned long long)K);
  };

  if (issuer) issue(0);

  v8f acc[4] = {};
  for (int kt = 0; kt < nk; ++kt) {
    if (issuer) {
      if (kt + 1 < nk) {
        issue(kt + 1);  // prefetch next pair into other buffer
        // in-order TENSORcnt: <=2 outstanding means the previous pair has landed
        __builtin_amdgcn_s_wait_tensorcnt(2);
      } else {
        __builtin_amdgcn_s_wait_tensorcnt(0);  // drain on last iteration
      }
    }
    __syncthreads();  // publish LDS tile to all waves

    const _Float16* As = &Abuf[kt & 1][0];
    const _Float16* Ws = &Wbuf[kt & 1][0];
    // A fragment (16x32): row = w*16+lh, two contiguous 8-half runs
    v8h alo = *(const v8h*)(As + (w * 16 + lh) * 32 + hi * 16);
    v8h ahi = *(const v8h*)(As + (w * 16 + lh) * 32 + hi * 16 + 8);
    v16h a;
#pragma unroll
    for (int t = 0; t < 8; ++t) { a[t] = alo[t]; a[t + 8] = ahi[t]; }
#pragma unroll
    for (int j = 0; j < 4; ++j) {
      v16h bmat = *(const v16h*)(Ws + (j * 16 + lh) * 32 + hi * 16);
      acc[j] = __builtin_amdgcn_wmma_f32_16x16x32_f16(
          false, a, false, bmat, (short)0, acc[j], false, false);
    }
    __syncthreads();  // compute done before buffer is overwritten next round
  }

#pragma unroll
  for (int j = 0; j < 4; ++j) {
#pragma unroll
    for (int r = 0; r < 8; ++r) {
      int row = m0 + w * 16 + r + hi * 8;
      int col = n0 + j * 16 + lh;
      if (Ch) Ch[(size_t)row * N + col] = (_Float16)acc[j][r];
      else    Cf[(size_t)row * N + col] = acc[j][r];
    }
  }
}

// ---------------------------------------------------------------- flash attention (causal), one wave = 16 queries of one (b,h)
__global__ __launch_bounds__(256) void flash_attn_kernel(
    const _Float16* __restrict__ qkv, _Float16* __restrict__ attn) {
  __shared__ _Float16 Pl[8][16][33];  // per-wave P staging, padded
  const int lane = threadIdx.x & 31;
  const int lh   = lane & 15;
  const int hi   = lane >> 4;
  const int w    = threadIdx.x >> 5;
  const int gw   = blockIdx.x * 8 + w;
  const int qt = gw & 127;            // S/16 = 128 query tiles
  const int h  = (gw >> 7) & 15;
  const int b  = gw >> 11;
  const int q0 = qt << 4;

  // Q fragments over hd=64 (two K=32 chunks), A layout
  v16h aq[2];
  {
    const _Float16* qrow = qkv + ((size_t)(b * SEQ + q0 + lh)) * N3 + h * HD;
#pragma unroll
    for (int c = 0; c < 2; ++c) {
      v8h lo  = *(const v8h*)(qrow + c * 32 + hi * 16);
      v8h hi8 = *(const v8h*)(qrow + c * 32 + hi * 16 + 8);
#pragma unroll
      for (int t = 0; t < 8; ++t) { aq[c][t] = lo[t]; aq[c][t + 8] = hi8[t]; }
    }
  }

  float mrow[8], lrow[8];
#pragma unroll
  for (int r = 0; r < 8; ++r) { mrow[r] = NEG_INF; lrow[r] = 0.0f; }
  v8f o[4] = {};
  const float scale = 0.125f;  // 1/sqrt(64)

  const int kend = q0 + 16;
  for (int k0 = 0; k0 < kend; k0 += 32) {
    // ---- scores S = Q K^T for 32 keys (two 16-key halves)
    v8f sc[2] = {};
#pragma unroll
    for (int kk = 0; kk < 2; ++kk) {
#pragma unroll
      for (int c = 0; c < 2; ++c) {
        const _Float16* kp = qkv +
            ((size_t)(b * SEQ + k0 + kk * 16 + lh)) * N3 + DM + h * HD +
            c * 32 + hi * 16;
        v16h bk = *(const v16h*)kp;  // contiguous along hd
        sc[kk] = __builtin_amdgcn_wmma_f32_16x16x32_f16(
            false, aq[c], false, bk, (short)0, sc[kk], false, false);
      }
    }
    // ---- online softmax (f32). Row M = r + hi*8 lives in 16 lanes of one half.
    float p0[8], p1[8], alpha[8];
#pragma unroll
    for (int r = 0; r < 8; ++r) {
      int qrow = q0 + r + hi * 8;
      float s0 = sc[0][r] * scale; if (k0 + lh      > qrow) s0 = NEG_INF;
      float s1 = sc[1][r] * scale; if (k0 + 16 + lh > qrow) s1 = NEG_INF;
      float tmax = fmaxf(s0, s1);
#pragma unroll
      for (int off = 8; off >= 1; off >>= 1)
        tmax = fmaxf(tmax, __shfl_xor(tmax, off, 32));
      float mnew = fmaxf(mrow[r], tmax);
      alpha[r] = __expf(mrow[r] - mnew);
      p0[r] = __expf(s0 - mnew);
      p1[r] = __expf(s1 - mnew);
      float psum = p0[r] + p1[r];
#pragma unroll
      for (int off = 8; off >= 1; off >>= 1)
        psum += __shfl_xor(psum, off, 32);
      lrow[r] = lrow[r] * alpha[r] + psum;
      mrow[r] = mnew;
    }
    // ---- C-layout -> A-layout transpose of P through LDS (same-wave DS is in-order;
    //      s_wait_dscnt 0 makes the cross-lane stores visible before loads)
#pragma unroll
    for (int r = 0; r < 8; ++r) {
      Pl[w][r + hi * 8][lh]      = (_Float16)p0[r];
      Pl[w][r + hi * 8][16 + lh] = (_Float16)p1[r];
    }
    asm volatile("s_wait_dscnt 0x0" ::: "memory");
    v16h aP;
#pragma unroll
    for (int t = 0; t < 16; ++t) {
      int col = hi * 16 + t;
      aP[t] = Pl[w][lh][col];
    }
    // ---- rescale O, accumulate P @ V (N=hd tiled by 16)
#pragma unroll
    for (int j = 0; j < 4; ++j) {
#pragma unroll
      for (int r = 0; r < 8; ++r) o[j][r] *= alpha[r];
    }
#pragma unroll
    for (int j = 0; j < 4; ++j) {
      const _Float16* vp = qkv +
          ((size_t)(b * SEQ + k0 + hi * 16)) * N3 + 2 * DM + h * HD +
          j * 16 + lh;
      v16h bv;
#pragma unroll
      for (int t = 0; t < 16; ++t) bv[t] = vp[(size_t)t * N3];
      o[j] = __builtin_amdgcn_wmma_f32_16x16x32_f16(
          false, aP, false, bv, (short)0, o[j], false, false);
    }
  }

  // ---- normalize and store attn output [B][S][H*hd] f16
#pragma unroll
  for (int j = 0; j < 4; ++j) {
#pragma unroll
    for (int r = 0; r < 8; ++r) {
      int row = q0 + r + hi * 8;
      attn[((size_t)(b * SEQ + row)) * DM + h * HD + j * 16 + lh] =
          (_Float16)(o[j][r] / lrow[r]);
    }
  }
}

// ---------------------------------------------------------------- launch
extern "C" void kernel_launch(void* const* d_in, const int* in_sizes, int n_in,
                              void* d_out, int out_size, void* d_ws, size_t ws_size,
                              hipStream_t stream) {
  const float* x    = (const float*)d_in[0];
  const int*   tpos = (const int*)d_in[1];
  const float* wqkv = (const float*)d_in[2];
  const float* wo   = (const float*)d_in[3];

  char* ws = (char*)d_ws;
  size_t off = 0;
  _Float16* xh    = (_Float16*)(ws + off); off += (size_t)BATCH * SEQ * DM * 2;
  _Float16* wqkvh = (_Float16*)(ws + off); off += (size_t)N3 * DM * 2;
  _Float16* woh   = (_Float16*)(ws + off); off += (size_t)DM * DM * 2;
  _Float16* qkvh  = (_Float16*)(ws + off); off += (size_t)BATCH * SEQ * N3 * 2;
  _Float16* attnh = (_Float16*)(ws + off);

  f32_to_f16_kernel<<<1024, 256, 0, stream>>>(x, xh, BATCH * SEQ * DM);
  f32_to_f16_kernel<<<512,  256, 0, stream>>>(wqkv, wqkvh, N3 * DM);
  f32_to_f16_kernel<<<256,  256, 0, stream>>>(wo, woh, DM * DM);

  // QKV projection: M=8192, N=3072, K=1024 -> (8192/128)*(3072/64) = 3072 blocks
  gemm_wmma_tdm<<<3072, 256, 0, stream>>>(xh, wqkvh, qkvh, nullptr,
                                          BATCH * SEQ, N3, DM);
  // RoPE on q,k sections
  rope_kernel<<<(BATCH * SEQ * NH * 32 * 2) / 256, 256, 0, stream>>>(qkvh, tpos);
  // Flash attention: 4*16*128 = 8192 waves
  flash_attn_kernel<<<1024, 256, 0, stream>>>(qkvh, attnh);
  // Output projection: M=8192, N=1024, K=1024 -> 1024 blocks, f32 out
  gemm_wmma_tdm<<<1024, 256, 0, stream>>>(attnh, woh, nullptr, (float*)d_out,
                                          BATCH * SEQ, DM, DM);
}
